// TransformField_21294447853816
// MI455X (gfx1250) — compile-verified, matrix operand order checked by reading, not run on hardware
//
#include <hip/hip_runtime.h>

typedef _Float16 f16;
typedef __attribute__((ext_vector_type(16))) _Float16 v16h;
typedef __attribute__((ext_vector_type(8)))  _Float16 v8h;
typedef __attribute__((ext_vector_type(8)))  float    v8f;

#define TILE_ROWS 128
#define NWAVES    8

// ---------------------------------------------------------------------------
// A-operand loader (CDNA5 16-bit A layout, wave32):
// lanes 0-15 = rows M=0..15 holding K = kb+{0..7,16..23},
// lanes 16-31 = same rows holding K = kb+{8..15,24..31}.
// Two 16-byte LDS loads per lane (ds_load_b128).
// ---------------------------------------------------------------------------
__device__ __forceinline__ v16h load_A_tile(const f16* A, int lda, int rowbase,
                                            int kbase, int lane) {
  int m    = lane & 15;
  int koff = (lane >> 4) << 3;  // 0 or 8
  const f16* p = A + (size_t)(rowbase + m) * lda + kbase + koff;
  v8h lo = *(const v8h*)p;
  v8h hi = *(const v8h*)(p + 16);
  v16h a;
#pragma unroll
  for (int i = 0; i < 8; ++i) { a[i] = lo[i]; a[8 + i] = hi[i]; }
  return a;
}

// ---------------------------------------------------------------------------
// Tile-GEMM layer with pre-packed f16 B operands.
// Packed layout: tile (kt,ct) at Wp[(kt*COLT + ct)*512 + lane*16 .. +15],
// zero-padded in both K and N -> B load is one unconditional 32-byte global
// load per k-step, reused across all row tiles held by the wave.
// COLT = NoutPad/16.  CW=min(COLT,8) column slots, NRT=CW row tiles per wave.
// ---------------------------------------------------------------------------
template <int COLT>
__device__ __forceinline__ void gemm_layer(
    const f16* A0, int lda0, int k0tiles,
    const f16* A1, int lda1, int k1tiles,   // optional split-K segment
    const f16* __restrict__ Wp, const float* __restrict__ biasp,
    f16* OutL, int ldo, float* OutG, int ldog, int nValid, int rowsValid,
    bool relu, int wave, int lane) {
  constexpr int CW  = (COLT < 8) ? COLT : 8;
  constexpr int NRT = CW;                      // row tiles per wave
  constexpr size_t WSTR = (size_t)COLT * 512;  // halves per kt row
  const int rtStart = (wave / CW) * NRT;
  for (int ct = (wave % CW); ct < COLT; ct += CW) {
    const f16* wp = Wp + (size_t)ct * 512 + (size_t)lane * 16;
    v8f acc[NRT];
#pragma unroll
    for (int r = 0; r < NRT; ++r) acc[r] = {};
    for (int kt = 0; kt < k0tiles; ++kt) {
      v16h b = *(const v16h*)wp;
      __builtin_prefetch((const void*)(wp + WSTR), 0, 3);
      wp += WSTR;
#pragma unroll
      for (int r = 0; r < NRT; ++r) {
        v16h a = load_A_tile(A0, lda0, (rtStart + r) << 4, kt * 32, lane);
        acc[r] = __builtin_amdgcn_wmma_f32_16x16x32_f16(
            false, a, false, b, (short)0, acc[r], false, false);
      }
    }
    for (int kt = 0; kt < k1tiles; ++kt) {
      v16h b = *(const v16h*)wp;
      __builtin_prefetch((const void*)(wp + WSTR), 0, 3);
      wp += WSTR;
#pragma unroll
      for (int r = 0; r < NRT; ++r) {
        v16h a = load_A_tile(A1, lda1, (rtStart + r) << 4, kt * 32, lane);
        acc[r] = __builtin_amdgcn_wmma_f32_16x16x32_f16(
            false, a, false, b, (short)0, acc[r], false, false);
      }
    }
    // Epilogue.  C/D layout: VGPR v -> M = v + (lane>=16 ? 8 : 0), N = lane&15.
    int n    = (ct << 4) + (lane & 15);
    float bv = biasp[n];
    int rsel = (lane >> 4) << 3;
#pragma unroll
    for (int r = 0; r < NRT; ++r) {
#pragma unroll
      for (int v = 0; v < 8; ++v) {
        float d = acc[r][v] + bv;
        if (relu) d = fmaxf(d, 0.0f);
        int row = ((rtStart + r) << 4) + rsel + v;
        if (OutG) {
          if (n < nValid && row < rowsValid) OutG[(size_t)row * ldog + n] = d;
        } else {
          OutL[(size_t)row * ldo + n] = (f16)d;
        }
      }
    }
  }
}

// ---------------------------------------------------------------------------
// Weight packer: f32 row-major [K x Nout] -> f16 WMMA-B tiles, zero-padded.
// One block packs one 16x32 tile (512 halves).  tile = kt*colTiles + ct.
// ---------------------------------------------------------------------------
__global__ __launch_bounds__(512) void pack_weight_kernel(
    const float* __restrict__ W, f16* __restrict__ Wp,
    int K, int Nout, int colTiles) {
  int tile = blockIdx.x;
  int kt = tile / colTiles, ct = tile - kt * colTiles;
  int t = threadIdx.x;          // 0..511
  int lane = t >> 4, i = t & 15;
  int n    = (ct << 4) + (lane & 15);
  int koff = (lane >> 4) << 3;
  int k    = kt * 32 + koff + (i & 7) + ((i >= 8) ? 16 : 0);
  float v = (k < K && n < Nout) ? W[(size_t)k * Nout + n] : 0.0f;
  Wp[(size_t)tile * 512 + (size_t)lane * 16 + i] = (f16)v;
}

__global__ void pack_bias_kernel(const float* __restrict__ b,
                                 float* __restrict__ bp, int Nout, int NoutPad) {
  int n = blockIdx.x * blockDim.x + threadIdx.x;
  if (n < NoutPad) bp[n] = (n < Nout) ? b[n] : 0.0f;
}

// ---------------------------------------------------------------------------
// Voxel branch via WMMA: 128 voxels per workgroup.
// vout[m] = relu(x[inds[m], :63] @ uw1 + ub1) @ uw2 + ub2
// Gather indices clamped (uniform loads); only the final store is row-guarded.
// ---------------------------------------------------------------------------
__global__ __launch_bounds__(256, 1) void voxel_wmma_kernel(
    const float* __restrict__ x, const int* __restrict__ inds,
    const f16* __restrict__ puw1, const float* __restrict__ pub1,
    const f16* __restrict__ puw2, const float* __restrict__ pub2,
    float* __restrict__ vout, int M) {
  extern __shared__ __align__(16) char smem_raw[];
  f16* vf = (f16*)smem_raw;   // 128 x  64 (63 + zero pad)
  f16* v1 = vf + 128 * 64;    // 128 x 128
  __shared__ int indbuf[128];

  const int tid = threadIdx.x, wave = tid >> 5, lane = tid & 31;
  const int row0 = blockIdx.x * 128;
  if (tid < 128) {
    int r = row0 + tid;
    indbuf[tid] = inds[(r < M) ? r : (M - 1)];
  }
  __syncthreads();
  for (int idx = tid; idx < 128 * 63; idx += 256) {
    int r = idx / 63, c = idx - r * 63;
    vf[r * 64 + c] = (f16)x[(size_t)indbuf[r] * 79 + c];
  }
  for (int r = tid; r < 128; r += 256) vf[r * 64 + 63] = (f16)0.0f;
  __syncthreads();
  gemm_layer<8>(vf, 64, 2, nullptr, 0, 0, puw1, pub1,
                v1, 128, nullptr, 0, 128, 128, true, wave, lane);
  __syncthreads();
  gemm_layer<4>(v1, 128, 4, nullptr, 0, 0, puw2, pub2,
                nullptr, 0, vout + (size_t)row0 * 64, 64, 64, M - row0,
                false, wave, lane);
}

// ---------------------------------------------------------------------------
// Fully fused per-point network: one workgroup = 128 points, 8 wave32,
// activations ping-pong in ~248 KB of dynamic LDS as f16.
// ---------------------------------------------------------------------------
__global__ __launch_bounds__(256, 1) void fused_field_kernel(
    const float* __restrict__ x, const int* __restrict__ inverse_map,
    const float* __restrict__ vout,
    const f16* __restrict__ pfw1, const float* __restrict__ pfb1,
    const f16* __restrict__ pfw2, const float* __restrict__ pfb2,
    const f16* __restrict__ pbw1, const float* __restrict__ pbb1,
    const f16* __restrict__ pbw2, const float* __restrict__ pbb2,
    const f16* __restrict__ pbw3, const float* __restrict__ pbb3,
    const f16* __restrict__ pnw0, const float* __restrict__ pnb0,
    const f16* __restrict__ pnw2, const float* __restrict__ pnb2,
    const f16* __restrict__ pnw4, const float* __restrict__ pnb4,
    const f16* __restrict__ pnw6, const float* __restrict__ pnb6,
    const f16* __restrict__ pnw8, const float* __restrict__ pnb8,
    float* __restrict__ out) {
  extern __shared__ __align__(16) char smem_raw[];
  f16* pe = (f16*)smem_raw;        // 128 x  64  (63 + zero pad)
  f16* fp = pe + 128 * 64;         // 128 x 128  ([front | pcd])
  f16* tA = fp + 128 * 128;        // 128 x  64
  f16* tB = tA + 128 * 64;         // 128 x  64
  f16* xx = tB + 128 * 64;         // 128 x 160  (76 feats + 64 back + 20 pad)
  f16* hA = xx + 128 * 160;        // 128 x 256
  f16* hB = hA + 128 * 256;        // 128 x 256
  __shared__ int imbuf[128];

  const int tid  = threadIdx.x;
  const int wave = tid >> 5;
  const int lane = tid & 31;
  const int row0 = blockIdx.x * TILE_ROWS;

  if (tid < 128) imbuf[tid] = inverse_map[row0 + tid];

  // ---- Stage inputs: unconditional loads + separate zero fills ----
  for (int idx = tid; idx < 128 * 63; idx += 256) {    // pe[:, :63]
    int r = idx / 63, c = idx - r * 63;
    pe[r * 64 + c] = (f16)x[(size_t)(row0 + r) * 79 + c];
  }
  for (int r = tid; r < 128; r += 256) pe[r * 64 + 63] = (f16)0.0f;
  for (int idx = tid; idx < 128 * 76; idx += 256) {    // xx[:, :76]
    int r = idx / 76, c = idx - r * 76;
    xx[r * 160 + c] = (f16)x[(size_t)(row0 + r) * 79 + c];
  }
  for (int idx = tid; idx < 128 * 84; idx += 256) {    // xx[:, 76:160] = 0
    int r = idx / 84, c = 76 + (idx - r * 84);
    xx[r * 160 + c] = (f16)0.0f;
  }
  __syncthreads();
  for (int idx = tid; idx < 128 * 64; idx += 256) {    // gather pcd -> fp[:,64:]
    int r = idx >> 6, c = idx & 63;
    fp[r * 128 + 64 + c] = (f16)vout[(size_t)imbuf[r] * 64 + c];
  }
  __syncthreads();

  // front = relu(pe@fw1+fb1)@fw2+fb2 -> fp[:, :64]
  gemm_layer<4>(pe, 64, 2, nullptr, 0, 0, pfw1, pfb1, tA, 64, nullptr, 0, 64, 128, true, wave, lane);
  __syncthreads();
  gemm_layer<4>(tA, 64, 2, nullptr, 0, 0, pfw2, pfb2, fp, 128, nullptr, 0, 64, 128, false, wave, lane);
  __syncthreads();
  // back MLP
  gemm_layer<4>(fp, 128, 4, nullptr, 0, 0, pbw1, pbb1, tA, 64, nullptr, 0, 64, 128, true, wave, lane);
  __syncthreads();
  gemm_layer<4>(tA, 64, 2, nullptr, 0, 0, pbw2, pbb2, tB, 64, nullptr, 0, 64, 128, true, wave, lane);
  __syncthreads();
  gemm_layer<4>(tB, 64, 2, nullptr, 0, 0, pbw3, pbb3, xx + 76, 160, nullptr, 0, 64, 128, false, wave, lane);
  __syncthreads();
  // h = relu(xx@nw0+nb0)  (K=140 padded to 160 -> 5 k-tiles)
  gemm_layer<16>(xx, 160, 5, nullptr, 0, 0, pnw0, pnb0, hA, 256, nullptr, 0, 256, 128, true, wave, lane);
  __syncthreads();
  gemm_layer<16>(hA, 256, 8, nullptr, 0, 0, pnw2, pnb2, hB, 256, nullptr, 0, 256, 128, true, wave, lane);
  __syncthreads();
  // h = relu([h|xx]@nw4+nb4): split-K, packed nw4 covers k-tiles 0..12
  gemm_layer<16>(hB, 256, 8, xx, 160, 5, pnw4, pnb4, hA, 256, nullptr, 0, 256, 128, true, wave, lane);
  __syncthreads();
  gemm_layer<16>(hA, 256, 8, nullptr, 0, 0, pnw6, pnb6, hB, 256, nullptr, 0, 256, 128, true, wave, lane);
  __syncthreads();
  // out = h@nw8+nb8  (Nout=12 padded to 16, guarded global store)
  gemm_layer<1>(hB, 256, 8, nullptr, 0, 0, pnw8, pnb8, nullptr, 0,
                out + (size_t)row0 * 12, 12, 12, 128, false, wave, lane);
}

__global__ void write_count_kernel(float* out, int pos, int M) {
  if (threadIdx.x == 0 && blockIdx.x == 0) out[pos] = (float)M;
}

// ---------------------------------------------------------------------------
extern "C" void kernel_launch(void* const* d_in, const int* in_sizes, int n_in,
                              void* d_out, int out_size, void* d_ws,
                              size_t ws_size, hipStream_t stream) {
  const float* x    = (const float*)d_in[0];
  const int*   inds = (const int*)d_in[1];
  const int*   inv  = (const int*)d_in[2];
  // 12 packed layers: fw1 fw2 bw1 bw2 bw3 nw0 nw2 nw4 nw6 nw8 uw1 uw2
  const float* W[12] = {
      (const float*)d_in[3],  (const float*)d_in[5],
      (const float*)d_in[11], (const float*)d_in[13], (const float*)d_in[15],
      (const float*)d_in[17], (const float*)d_in[19], (const float*)d_in[21],
      (const float*)d_in[23], (const float*)d_in[25],
      (const float*)d_in[7],  (const float*)d_in[9]};
  const float* B[12] = {
      (const float*)d_in[4],  (const float*)d_in[6],
      (const float*)d_in[12], (const float*)d_in[14], (const float*)d_in[16],
      (const float*)d_in[18], (const float*)d_in[20], (const float*)d_in[22],
      (const float*)d_in[24], (const float*)d_in[26],
      (const float*)d_in[8],  (const float*)d_in[10]};

  const int Ks[12] = {63, 64, 128, 64, 64, 140, 256, 396, 256, 256, 63, 128};
  const int Ns[12] = {64, 64, 64, 64, 64, 256, 256, 256, 256, 12, 128, 64};
  const int KT[12] = {2, 2, 4, 2, 2, 5, 8, 13, 8, 8, 2, 4};
  const int CT[12] = {4, 4, 4, 4, 4, 16, 16, 16, 16, 1, 8, 4};

  const int M    = in_sizes[1];
  const int Npts = in_sizes[0] / 79;
  float* out  = (float*)d_out;
  float* vout = (float*)d_ws;  // M x 64 f32

  size_t off = ((size_t)M * 64 * sizeof(float) + 1023) & ~(size_t)1023;
  f16*   Wp[12];
  float* bp[12];
  for (int l = 0; l < 12; ++l) {
    Wp[l] = (f16*)((char*)d_ws + off);
    off += ((size_t)KT[l] * CT[l] * 512 * sizeof(f16) + 1023) & ~(size_t)1023;
  }
  for (int l = 0; l < 12; ++l) {
    bp[l] = (float*)((char*)d_ws + off);
    off += 1024;
  }

  // 1) pack weights/biases into WMMA-B layout (deterministic, every call)
  for (int l = 0; l < 12; ++l) {
    pack_weight_kernel<<<KT[l] * CT[l], 512, 0, stream>>>(W[l], Wp[l], Ks[l], Ns[l], CT[l]);
    pack_bias_kernel<<<1, CT[l] * 16, 0, stream>>>(B[l], bp[l], Ns[l], CT[l] * 16);
  }

  // 2) voxel branch -> vout (WMMA, 128 voxels per workgroup)
  size_t vshm = (size_t)(128 * (64 + 128)) * sizeof(f16);
  voxel_wmma_kernel<<<(M + 127) / 128, 256, vshm, stream>>>(
      x, inds, Wp[10], bp[10], Wp[11], bp[11], vout, M);

  // 3) fused per-point network (WMMA)
  size_t shmem = (size_t)(128 * (64 + 128 + 64 + 64 + 160 + 256 + 256)) * sizeof(f16);
  fused_field_kernel<<<Npts / TILE_ROWS, 256, shmem, stream>>>(
      x, inv, vout,
      Wp[0], bp[0], Wp[1], bp[1], Wp[2], bp[2], Wp[3], bp[3], Wp[4], bp[4],
      Wp[5], bp[5], Wp[6], bp[6], Wp[7], bp[7], Wp[8], bp[8], Wp[9], bp[9],
      out);

  // 4) second tuple element: unique voxel count
  write_count_kernel<<<1, 1, 0, stream>>>(out, out_size - 1, M);
}
